// GraphNN_4587025072779
// MI455X (gfx1250) — matrix operand order repaired and checked by reading, not compiled, not run on hardware
//
#include <hip/hip_runtime.h>
#include <hip/hip_bf16.h>

#define NNODES 50000
#define FIN    256
#define HD     128
#define NHEADS 4
#define LAYERS 3
#define TYPES  3
#define NEDGES 250000
#define NEGS   0.2f

typedef __attribute__((ext_vector_type(16))) __bf16 v16bf;
typedef __attribute__((ext_vector_type(8)))  float  v8f;

// ---------- helpers ----------
__device__ __forceinline__ unsigned short f2bf(float f) {
    unsigned u = __float_as_uint(f);
    u += 0x7FFFu + ((u >> 16) & 1u);            // round-to-nearest-even
    return (unsigned short)(u >> 16);
}
__device__ __forceinline__ float bfu2f(unsigned short us) {
    return __uint_as_float(((unsigned)us) << 16);
}
__device__ __forceinline__ __bf16 usbf(unsigned short us) {
    return __builtin_bit_cast(__bf16, us);
}
// monotonic float <-> uint mapping for atomicMax-based segment max
__device__ __forceinline__ unsigned fmono(float f) {
    unsigned u = __float_as_uint(f);
    return u ^ ((unsigned)((int)u >> 31) | 0x80000000u);
}
__device__ __forceinline__ float funmono(unsigned u) {
    unsigned m = (u & 0x80000000u) ? (u ^ 0x80000000u) : ~u;
    return __uint_as_float(m);
}
__device__ __forceinline__ float leaky(float x) { return x > 0.f ? x : NEGS * x; }
__device__ __forceinline__ float wred(float v) {
#pragma unroll
    for (int off = 16; off > 0; off >>= 1) v += __shfl_xor(v, off, 32);
    return v;
}
// load 4 consecutive bf16 (8 bytes, aligned) -> 4 floats
__device__ __forceinline__ void ld4bf(const unsigned short* __restrict__ p, float f[4]) {
    const uint2 v = *(const uint2*)p;
    f[0] = bfu2f((unsigned short)(v.x & 0xffffu));
    f[1] = bfu2f((unsigned short)(v.x >> 16));
    f[2] = bfu2f((unsigned short)(v.y & 0xffffu));
    f[3] = bfu2f((unsigned short)(v.y >> 16));
}

// ---------- elementwise conversion kernels ----------
__global__ void k_f2bf(const float* __restrict__ src, unsigned short* __restrict__ dst, long n) {
    long i = (long)blockIdx.x * blockDim.x + threadIdx.x;
    if (i < n) dst[i] = f2bf(src[i]);
}

// pack Wl[lt,:,0:512] and Wr[lt,:,0:512] into Wbf[lt,128,1024] (cols 0..511 = Wl, 512..1023 = Wr)
__global__ void k_wcat(const float* __restrict__ Wl, const float* __restrict__ Wr,
                       unsigned short* __restrict__ Wbf, long n) {
    long i = (long)blockIdx.x * blockDim.x + threadIdx.x;
    if (i >= n) return;                         // n = L*T*128*512
    long lt  = i >> 16;                         // / (128*512)
    long rem = i & 65535;
    long row = rem >> 9;                        // / 512
    long col = rem & 511;
    unsigned short* base = Wbf + lt * 131072 + row * 1024;
    base[col]       = f2bf(Wl[i]);
    base[512 + col] = f2bf(Wr[i]);
}

__global__ void k_zero(unsigned int* __restrict__ p, long n) {
    long i = (long)blockIdx.x * blockDim.x + threadIdx.x;
    if (i < n) p[i] = 0u;
}

// ---------- WMMA bf16 GEMM: C[M,Nc] = A[M,K] * B[K,Nc], LDS-staged ----------
// blockDim = 256 (8 waves). Block computes rows [m0,m0+16) x cols [ntb*16, ntb*16+128).
// grid = (M/16, Nc/128). Both operands staged in LDS; B is pre-shuffled into
// fragment-ready order so each lane reads its 16 elements as one contiguous 32B run.
template <bool F32OUT, int KT>
__global__ __launch_bounds__(256) void k_gemm(const unsigned short* __restrict__ A,
                                              const unsigned short* __restrict__ B,
                                              float* __restrict__ outF,
                                              unsigned short* __restrict__ outB,
                                              int Nc) {
    constexpr int KSTEPS = KT / 32;
    __shared__ unsigned short sA[16][KT];
    __shared__ unsigned short sB[KSTEPS][8][32][16];

    const int tid = threadIdx.x;
    const int m0  = blockIdx.x * 16;
    const int ntb = blockIdx.y * 8;             // first n-tile of this block

    // fill sA (16 x KT), coalesced 8B chunks
    for (int i = tid; i < 16 * (KT / 4); i += 256) {
        const int r  = i / (KT / 4);
        const int c4 = (i % (KT / 4)) * 4;
        *(uint2*)&sA[r][c4] = *(const uint2*)(A + (size_t)(m0 + r) * KT + c4);
    }
    // fill sB: global reads row-major coalesced; scatter into fragment order on LDS side
    for (int i = tid; i < KT * 32; i += 256) {
        const int krow = i >> 5;                // 0..KT-1
        const int c4   = (i & 31) * 4;          // 0..124
        const uint2 v  = *(const uint2*)(B + (size_t)krow * Nc + ntb * 16 + c4);
        unsigned short u[4] = {
            (unsigned short)(v.x & 0xffffu), (unsigned short)(v.x >> 16),
            (unsigned short)(v.y & 0xffffu), (unsigned short)(v.y >> 16)
        };
        const int ks  = krow >> 5;
        const int rem = krow & 31;
        const int bhi = rem >> 4;
        const int be  = rem & 15;
#pragma unroll
        for (int j = 0; j < 4; ++j) {
            const int col = c4 + j;
            sB[ks][col >> 4][bhi * 16 + (col & 15)][be] = u[j];
        }
    }
    __syncthreads();

    const int wave = tid >> 5;
    const int lane = tid & 31;
    const int mr = lane & 15;
    const int hi = lane >> 4;

    v8f c = {};
#pragma unroll
    for (int ks = 0; ks < KSTEPS; ++ks) {
        v16bf a, b;
        const unsigned short* ap = &sA[mr][ks * 32 + hi * 16];
#pragma unroll
        for (int j = 0; j < 8; ++j) { a[j] = usbf(ap[j]); a[j + 8] = usbf(ap[8 + j]); }
        const unsigned short* bp = &sB[ks][wave][lane][0];
#pragma unroll
        for (int e = 0; e < 16; ++e) b[e] = usbf(bp[e]);
        c = __builtin_amdgcn_wmma_f32_16x16x32_bf16(false, a, false, b, (short)0, c, false, false);
    }

    const int nn = (ntb + wave) * 16 + mr;
#pragma unroll
    for (int r = 0; r < 8; ++r) {
        const int mm = m0 + r + hi * 8;
        if (F32OUT) outF[(size_t)mm * Nc + nn] = c[r];
        else        outB[(size_t)mm * Nc + nn] = f2bf(c[r]);
    }
}

// ---------- node fuse: h=leaky(LN(tmp+proj_b)); ne=l2norm(emb); x0=0.7h+0.5ne ----------
__global__ void k_nodefuse(const float* __restrict__ tmp, const float* __restrict__ emb,
                           const float* __restrict__ pb, const float* __restrict__ lg,
                           const float* __restrict__ lb,
                           float* __restrict__ x0, float* __restrict__ x,
                           unsigned short* __restrict__ xbf) {
    const int wave = threadIdx.x >> 5, lane = threadIdx.x & 31;
    const int n = blockIdx.x * 8 + wave;
    if (n >= NNODES) return;
    const float* row = tmp + (size_t)n * HD;
    const float* er  = emb + (size_t)n * HD;
    float v[4], e[4];
    float s = 0.f, ss = 0.f;
#pragma unroll
    for (int j = 0; j < 4; ++j) {
        const int d = lane * 4 + j;
        v[j] = row[d] + pb[d];  s += v[j];
        e[j] = er[d];           ss += e[j] * e[j];
    }
    const float mu = wred(s) * (1.f / HD);
    float q = 0.f;
#pragma unroll
    for (int j = 0; j < 4; ++j) { const float dd = v[j] - mu; q += dd * dd; }
    const float rs  = rsqrtf(wred(q) * (1.f / HD) + 1e-5f);
    const float inv = 1.f / fmaxf(sqrtf(wred(ss)), 1e-12f);
#pragma unroll
    for (int j = 0; j < 4; ++j) {
        const int d = lane * 4 + j;
        float hval = leaky((v[j] - mu) * rs * lg[d] + lb[d]);
        float x0v  = 0.7f * hval + 0.5f * e[j] * inv;
        x0[(size_t)n * HD + d] = x0v;
        x [(size_t)n * HD + d] = x0v;
        xbf[(size_t)n * HD + d] = f2bf(x0v);
    }
}

// ---------- edge pass 1: e[E,4] = sum_d a*leaky(xl[src]+xr[dst]); segment max ----------
__global__ void k_edge_score(const unsigned short* __restrict__ xlxr,
                             const float* __restrict__ att,
                             const int* __restrict__ src, const int* __restrict__ dst,
                             float* __restrict__ esc, unsigned int* __restrict__ emaxu) {
    __shared__ float sat[NHEADS * HD];
    for (int j = threadIdx.x; j < NHEADS * HD; j += blockDim.x) sat[j] = att[j];
    __syncthreads();
    const int wave = threadIdx.x >> 5, lane = threadIdx.x & 31;
    const int e = blockIdx.x * 8 + wave;
    if (e >= NEDGES) return;
    const int s = src[e], d = dst[e];
    const unsigned short* xl = xlxr + (size_t)s * 1024;
    const unsigned short* xr = xlxr + (size_t)d * 1024 + 512;
    float hs[NHEADS];
    const int db = lane * 4;
#pragma unroll
    for (int h = 0; h < NHEADS; ++h) {
        float fl[4], fr[4];
        ld4bf(xl + h * HD + db, fl);
        ld4bf(xr + h * HD + db, fr);
        float acc = 0.f;
#pragma unroll
        for (int j = 0; j < 4; ++j)
            acc += sat[h * HD + db + j] * leaky(fl[j] + fr[j]);
        hs[h] = acc;
    }
#pragma unroll
    for (int h = 0; h < NHEADS; ++h) hs[h] = wred(hs[h]);
    if (lane == 0)
#pragma unroll
        for (int h = 0; h < NHEADS; ++h) {
            esc[(size_t)e * 4 + h] = hs[h];
            atomicMax(&emaxu[(size_t)d * 4 + h], fmono(hs[h]));
        }
}

// ---------- edge pass 2: ex = exp(e - max); denominator ----------
__global__ void k_edge_soft(float* __restrict__ esc, const unsigned int* __restrict__ emaxu,
                            float* __restrict__ den, const int* __restrict__ dst) {
    const long i = (long)blockIdx.x * blockDim.x + threadIdx.x;
    if (i >= (long)NEDGES * 4) return;
    const int e = (int)(i >> 2), h = (int)(i & 3);
    const int d = dst[e];
    const float ex = __expf(esc[i] - funmono(emaxu[(size_t)d * 4 + h]));
    esc[i] = ex;
    atomicAdd(&den[(size_t)d * 4 + h], ex);
}

// ---------- edge pass 3: out[dst] += (1/4) sum_h alpha_h * xl[src,h,:] ----------
__global__ void k_edge_aggr(const unsigned short* __restrict__ xlxr,
                            const float* __restrict__ esc, const float* __restrict__ den,
                            const int* __restrict__ src, const int* __restrict__ dst,
                            float* __restrict__ acc) {
    const int wave = threadIdx.x >> 5, lane = threadIdx.x & 31;
    const int e = blockIdx.x * 8 + wave;
    if (e >= NEDGES) return;
    const int s = src[e], d = dst[e];
    float al[NHEADS];
#pragma unroll
    for (int h = 0; h < NHEADS; ++h)
        al[h] = 0.25f * esc[(size_t)e * 4 + h] / (den[(size_t)d * 4 + h] + 1e-16f);
    const unsigned short* xl = xlxr + (size_t)s * 1024;
    const int db = lane * 4;
    float a[4] = {0.f, 0.f, 0.f, 0.f};
#pragma unroll
    for (int h = 0; h < NHEADS; ++h) {
        float fl[4];
        ld4bf(xl + h * HD + db, fl);
#pragma unroll
        for (int j = 0; j < 4; ++j) a[j] += al[h] * fl[j];
    }
#pragma unroll
    for (int j = 0; j < 4; ++j)
        atomicAdd(&acc[(size_t)d * HD + db + j], a[j]);
}

// ---------- layer finalize: x = leaky((acc + sum_t bias)/T) + sigmoid(skip)*x0 ----------
__global__ void k_layer_fin(const float* __restrict__ acc, const float* __restrict__ bias_i,
                            const float* __restrict__ skw, int li,
                            const float* __restrict__ x0,
                            float* __restrict__ xout, unsigned short* __restrict__ xbf) {
    const long i = (long)blockIdx.x * blockDim.x + threadIdx.x;
    if (i >= (long)NNODES * HD) return;
    const int d = (int)(i & (HD - 1));
    const float bsum = bias_i[d] + bias_i[HD + d] + bias_i[2 * HD + d];
    const float xu = (acc[i] + bsum) * (1.f / (float)TYPES);
    const float sg = 1.f / (1.f + __expf(-skw[li]));
    const float xn = leaky(xu) + sg * x0[i];
    xout[i] = xn;
    xbf[i]  = f2bf(xn);
}

// ---------- host ----------
extern "C" void kernel_launch(void* const* d_in, const int* in_sizes, int n_in,
                              void* d_out, int out_size, void* d_ws, size_t ws_size,
                              hipStream_t stream) {
    (void)in_sizes; (void)n_in; (void)out_size; (void)ws_size;
    const float* feat  = (const float*)d_in[0];
    const float* emb   = (const float*)d_in[1];
    const float* projw = (const float*)d_in[2];
    const float* projb = (const float*)d_in[3];
    const float* lng   = (const float*)d_in[4];
    const float* lnb   = (const float*)d_in[5];
    const float* skw   = (const float*)d_in[6];
    const float* Wl    = (const float*)d_in[7];
    const float* Wr    = (const float*)d_in[8];
    const float* att   = (const float*)d_in[9];
    const float* bias  = (const float*)d_in[10];
    const int*   esrc  = (const int*)d_in[11];
    const int*   edst  = (const int*)d_in[12];

    // carve workspace
    size_t off = 0;
    char* base = (char*)d_ws;
    auto alloc = [&](size_t bytes) -> void* {
        void* p = base + off;
        off = (off + bytes + 255) & ~(size_t)255;
        return p;
    };
    unsigned short* xlxr   = (unsigned short*)alloc((size_t)NNODES * 1024 * 2); // 102.4 MB
    unsigned short* featbf = xlxr;             // alias: only live before first xlxr GEMM
    unsigned short* pwbf   = (unsigned short*)alloc((size_t)FIN * HD * 2);
    unsigned short* Wbf    = (unsigned short*)alloc((size_t)LAYERS * TYPES * HD * 1024 * 2);
    float*          acc    = (float*)alloc((size_t)NNODES * HD * 4);
    float*          tmp    = acc;              // alias: GEMM1 output consumed before layers
    float*          x0     = (float*)alloc((size_t)NNODES * HD * 4);
    float*          x      = (float*)alloc((size_t)NNODES * HD * 4);
    unsigned short* xbf    = (unsigned short*)alloc((size_t)NNODES * HD * 2);
    float*          esc    = (float*)alloc((size_t)NEDGES * 4 * 4);
    unsigned int*   emaxu  = (unsigned int*)alloc((size_t)NNODES * 4 * 4);
    float*          den    = (float*)alloc((size_t)NNODES * 4 * 4);

    const int TPB = 256;
    auto blks = [](long n, int t) { return (unsigned)((n + t - 1) / t); };

    // conversions to bf16
    k_f2bf<<<blks((long)NNODES * FIN, TPB), TPB, 0, stream>>>(feat, featbf, (long)NNODES * FIN);
    k_f2bf<<<blks((long)FIN * HD, TPB), TPB, 0, stream>>>(projw, pwbf, (long)FIN * HD);
    k_wcat<<<blks((long)LAYERS * TYPES * HD * 512, TPB), TPB, 0, stream>>>(
        Wl, Wr, Wbf, (long)LAYERS * TYPES * HD * 512);

    // GEMM1: tmp[N,128] = feat[N,256] @ proj_w[256,128]  (WMMA bf16, K=256)
    {
        dim3 g(NNODES / 16, 1);
        k_gemm<true, FIN><<<g, TPB, 0, stream>>>(featbf, pwbf, tmp, nullptr, HD);
    }
    // node fuse -> x0, x, xbf
    k_nodefuse<<<NNODES / 8, TPB, 0, stream>>>(tmp, emb, projb, lng, lnb, x0, x, xbf);

    for (int li = 0; li < LAYERS; ++li) {
        k_zero<<<blks((long)NNODES * HD, TPB), TPB, 0, stream>>>((unsigned int*)acc, (long)NNODES * HD);
        for (int t = 0; t < TYPES; ++t) {
            // xlxr[N,1024] = x[N,128] @ [Wl|Wr][128,1024]  (WMMA bf16, K=128)
            dim3 g(NNODES / 16, 1024 / 128);
            k_gemm<false, HD><<<g, TPB, 0, stream>>>(
                xbf, Wbf + (size_t)(li * TYPES + t) * HD * 1024, nullptr, xlxr, 1024);
            k_zero<<<blks((long)NNODES * 4, TPB), TPB, 0, stream>>>(emaxu, (long)NNODES * 4);
            k_zero<<<blks((long)NNODES * 4, TPB), TPB, 0, stream>>>((unsigned int*)den, (long)NNODES * 4);
            const int* s = esrc + (size_t)t * NEDGES;
            const int* d = edst + (size_t)t * NEDGES;
            k_edge_score<<<NEDGES / 8, TPB, 0, stream>>>(
                xlxr, att + (size_t)(li * TYPES + t) * NHEADS * HD, s, d, esc, emaxu);
            k_edge_soft<<<blks((long)NEDGES * 4, TPB), TPB, 0, stream>>>(esc, emaxu, den, d);
            k_edge_aggr<<<NEDGES / 8, TPB, 0, stream>>>(xlxr, esc, den, s, d, acc);
        }
        float* xout = (li == LAYERS - 1) ? (float*)d_out : x;
        k_layer_fin<<<blks((long)NNODES * HD, TPB), TPB, 0, stream>>>(
            acc, bias + (size_t)li * TYPES * HD, skw, li, x0, xout, xbf);
    }
}
